// IrrepsLinear_64742337020475
// MI455X (gfx1250) — matrix operand order repaired and checked by reading, not compiled, not run on hardware
//
#include <hip/hip_runtime.h>
#include <hip/hip_bf16.h>
#include <stdint.h>

// ---------------------------------------------------------------------------
// Irreps linear (256x0e + 128x1o + 64x2e) as three GEMMs on CDNA5 WMMA.
// Precision: fp32 emulated via bf16x3 split (A = Ahi + Alo exactly;
// acc += Ahi*Blo + Alo*Bhi + Ahi*Bhi in f32 accumulators) on
// v_wmma_f32_16x16x32_bf16 — ~2^-16 rel err, ~3x fp32-WMMA throughput.
// Every x element is read exactly once and every out element written once
// (768 MB total -> ~33us at 23.3 TB/s HBM roofline); streaming traffic is
// non-temporal so L2/WGP$ stay dedicated to the packed-W fragments.
// B-fragment and A loads are software-pipelined one step ahead so the
// hardware waits are partial (loadcnt<=4) instead of full barriers.
// ---------------------------------------------------------------------------

typedef __attribute__((ext_vector_type(16))) __bf16        v16bf;
typedef __attribute__((ext_vector_type(8)))  float         v8f;
typedef __attribute__((ext_vector_type(8)))  unsigned int  v8u;

__device__ __forceinline__ unsigned as_u(float f)   { return __float_as_uint(f); }
__device__ __forceinline__ float    as_f(unsigned u){ return __uint_as_float(u); }

__device__ __forceinline__ v16bf frag(const v8u& v) {
    return __builtin_bit_cast(v16bf, v);
}

// hi dword = packed bf16 truncations of (x0, x1): elem0 in [15:0], elem1 in [31:16]
__device__ __forceinline__ unsigned pack_hi(float x0, float x1) {
    return (as_u(x1) & 0xffff0000u) | (as_u(x0) >> 16);
}
// exact split: x = hi + lo_f32 ; lo packed as truncated bf16 (correction term)
__device__ __forceinline__ void split_pair(float x0, float x1,
                                           unsigned& hi, unsigned& lo) {
    unsigned b0 = as_u(x0), b1 = as_u(x1);
    hi = (b1 & 0xffff0000u) | (b0 >> 16);
    float l0 = x0 - as_f(b0 & 0xffff0000u);
    float l1 = x1 - as_f(b1 & 0xffff0000u);
    lo = (as_u(l1) & 0xffff0000u) | (as_u(l0) >> 16);
}

// ---------------------------------------------------------------------------
// Prepass: pack W (row-major [K=i][N=o], K=N=MUL) into the exact B-fragment
// layout of V_WMMA_F32_16X16X32_BF16, split into hi (dwords 0..7) and lo
// (dwords 8..15) per (col-tile ct, k-tile kt, lane).
//   B frag: VGPR j, lane L: element K = kt*32 + (j>>2)*16 + (L>=16)*8 + 2(j&3) + {0,1},
//           N = ct*16 + (L&15)
// dst index = (((ct*KT + kt)*32 + lane)*16 + sel*8 + j)
// ---------------------------------------------------------------------------
template <int MUL>
__global__ __launch_bounds__(256)
void pack_w_kernel(const float* __restrict__ W, unsigned* __restrict__ dst) {
    constexpr int KT  = MUL / 32;
    constexpr int CT  = MUL / 16;
    constexpr int TOT = CT * KT * 32 * 16;
    const int tid = blockIdx.x * 256 + threadIdx.x;
    if (tid >= TOT) return;
    const int j    =  tid       & 7;
    const int sel  = (tid >> 3) & 1;
    const int lane = (tid >> 4) & 31;
    const int ctkt =  tid >> 9;
    const int kt   = ctkt % KT;
    const int ct   = ctkt / KT;
    const int k0   = kt * 32 + ((j >> 2) * 16) + ((lane >> 4) * 8) + 2 * (j & 3);
    const int nn   = ct * 16 + (lane & 15);
    const float w0 = W[(size_t)k0       * MUL + nn];
    const float w1 = W[(size_t)(k0 + 1) * MUL + nn];
    unsigned d;
    if (sel == 0) {
        d = pack_hi(w0, w1);
    } else {
        unsigned h, l;
        split_pair(w0, w1, h, l);
        d = l;
    }
    dst[tid] = d;
}

// ---------------------------------------------------------------------------
// Main GEMM: logical X'[n_rows, MUL] @ W[MUL, MUL], where
//   X'[r, i]  = x[n*960 + off + i*DIM + m],  r = n*DIM + m
//   out[n*960 + off + o*DIM + m] = scale * sum_i X'[r,i] W[i,o]
// One wave computes a 16-row stripe x NT col-tiles (16 cols each).
// ---------------------------------------------------------------------------
template <int MUL, int DIM, int NT>
__global__ __launch_bounds__(256)
void irreps_wmma_kernel(const float* __restrict__ x,
                        const unsigned* __restrict__ wfrag,
                        float* __restrict__ out,
                        int n_rows, int off, float scale) {
    constexpr int KT = MUL / 32;   // k-tiles of 32
    constexpr int CT = MUL / 16;   // col tiles of 16
    constexpr int CG = CT / NT;    // col groups per row stripe
    static_assert(CG * NT == CT, "NT must divide CT");

    const int lane = threadIdx.x & 31;
    const int wave = blockIdx.x * (blockDim.x >> 5) + (threadIdx.x >> 5);
    const int total = (n_rows / 16) * CG;
    if (wave >= total) return;          // wave-uniform: EXEC stays all-1s

    const int cg   = wave % CG;
    const int rt   = wave / CG;
    const int half = lane >> 4;
    const int ln   = lane & 15;

    // This lane's A row (M = ln), strided view into x
    const int r = rt * 16 + ln;
    const int n = r / DIM;
    const int m = r % DIM;
    const float* __restrict__ xrow = x + (size_t)n * 960 + off + m;

    // raw A for k-step kt: g0[e] = X'[r, kt*32 + half*8 + e],
    //                      g1[e] = X'[r, kt*32 + 16 + half*8 + e], e = 0..7
    auto loadA = [&](int kt, v8f& g0, v8f& g1) {
        if constexpr (DIM == 1) {
            g0 = __builtin_nontemporal_load(
                (const v8f*)(xrow + kt * 32 + half * 8));
            g1 = __builtin_nontemporal_load(
                (const v8f*)(xrow + kt * 32 + 16 + half * 8));
        } else {
#pragma unroll
            for (int e = 0; e < 8; ++e) {
                g0[e] = __builtin_nontemporal_load(
                    xrow + (size_t)(kt * 32 + half * 8 + e) * DIM);
                g1[e] = __builtin_nontemporal_load(
                    xrow + (size_t)(kt * 32 + 16 + half * 8 + e) * DIM);
            }
        }
    };
    // pre-packed B fragment (hi dwords 0..7, lo dwords 8..15)
    auto loadB = [&](int kt, int t, v8u& bh, v8u& bl) {
        const v8u* bp = (const v8u*)(wfrag +
            (((size_t)((cg * NT + t) * KT + kt) * 32 + lane) * 16));
        bh = bp[0];
        bl = bp[1];
    };
    // raw A -> bf16 fragments (A 16x32 layout: pairs (2j,2j+1) per dword)
    auto splitA = [&](const v8f& g0, const v8f& g1, v16bf& Ah, v16bf& Al) {
        v8u ahi, alo;
#pragma unroll
        for (int j = 0; j < 4; ++j) {
            unsigned h, l;
            split_pair(g0[2 * j], g0[2 * j + 1], h, l);
            ahi[j] = h; alo[j] = l;
            split_pair(g1[2 * j], g1[2 * j + 1], h, l);
            ahi[j + 4] = h; alo[j + 4] = l;
        }
        Ah = __builtin_bit_cast(v16bf, ahi);
        Al = __builtin_bit_cast(v16bf, alo);
    };

    v8f acc[NT];
#pragma unroll
    for (int t = 0; t < NT; ++t) { v8f z = {}; acc[t] = z; }

    // ---- software pipeline: A and B loaded one step ahead -----------------
    v8f ra0, ra1;                 // raw A, current kt
    v8u bh, bl, bh_n, bl_n;       // B fragment, current / next
    loadA(0, ra0, ra1);
    loadB(0, 0, bh, bl);

    for (int kt = 0; kt < KT; ++kt) {
        const int ktn = (kt + 1 < KT) ? kt + 1 : kt;  // uniform clamp, no branch

        v16bf Ah, Al;
        splitA(ra0, ra1, Ah, Al);   // VALU, overlaps outstanding loads
        v8f na0, na1;
        loadA(ktn, na0, na1);       // prefetch next k-step's A (HBM)

#pragma unroll
        for (int t = 0; t < NT; ++t) {
            // prefetch next tile's B before consuming the current one
            if (t + 1 < NT) loadB(kt, t + 1, bh_n, bl_n);
            else            loadB(ktn, 0,   bh_n, bl_n);

            // correction terms first, main term last
            acc[t] = __builtin_amdgcn_wmma_f32_16x16x32_bf16(
                false, Ah, false, frag(bl), (short)0, acc[t], false, false);
            acc[t] = __builtin_amdgcn_wmma_f32_16x16x32_bf16(
                false, Al, false, frag(bh), (short)0, acc[t], false, false);
            acc[t] = __builtin_amdgcn_wmma_f32_16x16x32_bf16(
                false, Ah, false, frag(bh), (short)0, acc[t], false, false);

            bh = bh_n;
            bl = bl_n;
        }
        ra0 = na0;
        ra1 = na1;
    }

    // D layout: VGPR rv, lane L -> M = rv + 8*(L>=16), N = L&15
    // Streaming output: non-temporal stores (written once, never re-read).
#pragma unroll
    for (int t = 0; t < NT; ++t) {
        const int col = (cg * NT + t) * 16 + ln;
#pragma unroll
        for (int rv = 0; rv < 8; ++rv) {
            const int rr = rt * 16 + rv + 8 * half;
            const int nn = rr / DIM;
            const int mm = rr % DIM;
            __builtin_nontemporal_store(
                acc[t][rv] * scale,
                out + (size_t)nn * 960 + off + col * DIM + mm);
        }
    }
}

// ---------------------------------------------------------------------------
extern "C" void kernel_launch(void* const* d_in, const int* in_sizes, int n_in,
                              void* d_out, int out_size, void* d_ws, size_t ws_size,
                              hipStream_t stream) {
    (void)n_in; (void)out_size; (void)ws_size;
    const float* x  = (const float*)d_in[0];
    const float* W0 = (const float*)d_in[1];
    const float* W1 = (const float*)d_in[2];
    const float* W2 = (const float*)d_in[3];
    float* out = (float*)d_out;
    const int N = in_sizes[0] / 960;   // 100000

    // packed-weight workspace (dwords): 65536 + 16384 + 4096 = 336 KB
    unsigned* ws0 = (unsigned*)d_ws;
    unsigned* ws1 = ws0 + 65536;
    unsigned* ws2 = ws0 + 65536 + 16384;

    pack_w_kernel<256><<<65536 / 256, 256, 0, stream>>>(W0, ws0);
    pack_w_kernel<128><<<16384 / 256, 256, 0, stream>>>(W1, ws1);
    pack_w_kernel< 64><<< 4096 / 256, 256, 0, stream>>>(W2, ws2);

    // block0: rows=N, MUL=256, NT=16 -> CG=1 (x read exactly once)
    {
        const int rows = N;
        const int waves = (rows / 16) * 1;
        const int blocks = (waves + 7) / 8;
        irreps_wmma_kernel<256, 1, 16><<<blocks, 256, 0, stream>>>(
            x, ws0, out, rows, 0, 0.0625f);                 // 1/sqrt(256)
    }
    // block1: rows=3N, MUL=128, NT=8 -> CG=1
    {
        const int rows = N * 3;
        const int waves = (rows / 16) * 1;
        const int blocks = (waves + 7) / 8;
        irreps_wmma_kernel<128, 3, 8><<<blocks, 256, 0, stream>>>(
            x, ws1, out, rows, 256, 0.08838834764831845f);  // 1/sqrt(128)
    }
    // block2: rows=5N, MUL=64, NT=4 -> CG=1
    {
        const int rows = N * 5;
        const int waves = (rows / 16) * 1;
        const int blocks = (waves + 7) / 8;
        irreps_wmma_kernel<64, 5, 4><<<blocks, 256, 0, stream>>>(
            x, ws2, out, rows, 640, 0.125f);                // 1/sqrt(64)
    }
}
